// GeneratorDiffWaveForTesting_34531537060386
// MI455X (gfx1250) — compile-verified, matrix operand order checked by reading, not run on hardware
//
#include <hip/hip_runtime.h>
#include <hip/hip_bf16.h>

// ---------------------------------------------------------------------------
// DiffWave-style generator forward for MI455X (gfx1250, wave32, WMMA + TDM).
// All channel-mixing convolutions are 16x16x32 f16 WMMA GEMMs; the mel
// conditioner tile is staged into LDS by the Tensor Data Mover.
// ---------------------------------------------------------------------------

typedef __attribute__((ext_vector_type(16))) _Float16 v16h;
typedef __attribute__((ext_vector_type(8)))  _Float16 v8h;
typedef __attribute__((ext_vector_type(8)))  float    v8f;
typedef __attribute__((ext_vector_type(4)))  unsigned int v4u;
typedef __attribute__((ext_vector_type(8)))  int      v8i;
typedef __attribute__((ext_vector_type(4)))  int      v4i;

#define TT   16384
#define BB   8
#define CC   64
#define DD   30
#define MELC 80
#define MELP 96   // mel channels padded to 3 K-chunks of 32

// ---------------------------------------------------------------------------
// WMMA fragment loader for A (M x K row-major) or B ([n][k] in LDS),
// K contiguous. CDNA5 16-bit A/B layout: lane L holds row/col (L & 15);
// kb = ((L>>4)<<3); elems 0..7 -> K = kb..kb+7, elems 8..15 -> kb+16..kb+23.
// ---------------------------------------------------------------------------
__device__ __forceinline__ v16h load_frag(const _Float16* __restrict__ base,
                                          int ld, int rbase, int kchunk, int lane) {
  int r  = rbase + (lane & 15);
  int kb = kchunk + ((lane >> 4) << 3);
  const _Float16* p = base + (size_t)r * ld + kb;
  union { v16h v; struct { v8h lo; v8h hi; } s; } u;
  u.s.lo = *(const v8h*)(p);        // global_load_b128 / ds_load_b128
  u.s.hi = *(const v8h*)(p + 16);
  return u.v;
}

#define WMMA_F16(A, B, C) \
  __builtin_amdgcn_wmma_f32_16x16x32_f16(false, (A), false, (B), (short)0, (C), false, false)

__device__ __forceinline__ float leaky04(float v) { return v >= 0.f ? v : 0.4f * v; }

// ---------------------------------------------------------------------------
// Residual layer kernel. Grid: (TT/64, BB), block 256 (8 waves).
// Wave w owns out-channel tile [16w, 16w+16); loops 4 time tiles of 16.
// ---------------------------------------------------------------------------
struct SMemA {                       // staging (f16 activations, [n][k])
  _Float16 bdil[3][64][80];          // 3 dilation taps of (h + t_now)
  _Float16 bmel[64][MELP];           // mel tile (TDM-filled; [t][k] contiguous)
};
struct SMemB {                       // post-GEMM
  float    z[128][66];               // z[oc][n] (gate | sig)
  _Float16 bspl[64][80];             // gated activations, [n][k]
};
union alignas(16) SMemU { SMemA a; SMemB b; };

__global__ __launch_bounds__(256)
void layer_kernel(const float* __restrict__ h_in, float* __restrict__ h_out,
                  float* __restrict__ skip, const _Float16* __restrict__ m_bf,
                  const float* __restrict__ t_now,        // [B][64]
                  const _Float16* __restrict__ wdil,      // [3][128][64]
                  const _Float16* __restrict__ wmel,      // [128][96]
                  const _Float16* __restrict__ wspl,      // [128][64]
                  const float* __restrict__ bz,           // [128] dil_b + melc_b
                  const float* __restrict__ spl_b,        // [128]
                  int d)
{
  __shared__ SMemU sm;
  const int b    = blockIdx.y;
  const int t0   = blockIdx.x * 64;
  const int tid  = threadIdx.x;
  const int wave = tid >> 5;
  const int lane = tid & 31;

  // ---- TDM: DMA the contiguous 64x96 f16 mel tile (12KB) into LDS ---------
  // m_bf layout is [b][t][96] so the tile at (b, t0) is one contiguous block.
  // 1D descriptor: data_size=8B, tile_dim0 = 12288/8 = 1536 units.
  if (wave == 0) {
    const _Float16* gsrc = m_bf + ((size_t)b * TT + t0) * MELP;
    unsigned long long gaddr = (unsigned long long)(uintptr_t)gsrc;
    unsigned lds_addr = (unsigned)(uintptr_t)(&sm.a.bmel[0][0]);
    v4u g0;
    g0[0] = 1u;                                        // count=1, user mode
    g0[1] = lds_addr;                                  // LDS byte address
    g0[2] = (unsigned)(gaddr & 0xffffffffu);           // global_addr[31:0]
    g0[3] = (unsigned)((gaddr >> 32) & 0x01ffffffu)    // global_addr[56:32]
          | (2u << 30);                                // type = 2 ("image")
    v8i g1;
    g1[0] = (3 << 16);        // workgroup_mask=0, data_size=3 (8 bytes)
    g1[1] = (1536 << 16);     // tensor_dim0[15:0] in [31:16]
    g1[2] = (1 << 16);        // tensor_dim0 hi = 0; tensor_dim1 = 1
    g1[3] = (1536 << 16);     // tensor_dim1 hi = 0; tile_dim0 = 1536
    g1[4] = 1;                // tile_dim1 = 1, tile_dim2 = 0
    g1[5] = 1536;             // tensor_dim0_stride[31:0]
    g1[6] = 0;                // stride hi, tensor_dim1_stride lo
    g1[7] = 0;
    v4i g2 = {0, 0, 0, 0};
    v4i g3 = {0, 0, 0, 0};
    v8i g4 = {0, 0, 0, 0, 0, 0, 0, 0};   // trailing group (clang-23 6-arg form)
    __builtin_amdgcn_tensor_load_to_lds(g0, g1, g2, g3, g4, 0);
  }

  // ---- prefetch the skip rows this block will read-modify-write -----------
  __builtin_prefetch(skip + ((size_t)b * CC + (tid & 63)) * TT + t0, 1, 3);

  // ---- stage dilation taps of (h + t_now) as f16, [tap][n][k] -------------
  for (int idx = tid; idx < 3 * 64 * 64; idx += 256) {
    int tap = idx >> 12;
    int rem = idx & 4095;
    int k = rem >> 6, n = rem & 63;
    int t = t0 + n + (tap - 1) * d;
    float v = 0.f;
    if (t >= 0 && t < TT)
      v = h_in[((size_t)b * CC + k) * TT + t] + t_now[b * CC + k];
    sm.a.bdil[tap][n][k] = (_Float16)v;
  }
  __builtin_amdgcn_s_wait_tensorcnt(0);   // TDM tile landed (no-op for waves 1..7)
  __syncthreads();

  const int mbase = wave * 16;

  // ---- A fragments (f16 weights, L2-resident) -----------------------------
  v16h a_dil[3][2], a_mel[3];
#pragma unroll
  for (int tap = 0; tap < 3; ++tap)
#pragma unroll
    for (int kc = 0; kc < 2; ++kc)
      a_dil[tap][kc] = load_frag(wdil + (size_t)tap * 128 * 64, 64, mbase, kc * 32, lane);
#pragma unroll
  for (int kc = 0; kc < 3; ++kc)
    a_mel[kc] = load_frag(wmel, MELP, mbase, kc * 32, lane);

  // ---- z GEMM. For each (tap,kc): 4 B-fragment loads, then 4 independent
  // WMMAs (distinct accumulators -> no RAW hazard, one dscnt wait per group).
  v8f accz[4];
#pragma unroll
  for (int j = 0; j < 4; ++j)
    accz[j] = (v8f){0.f, 0.f, 0.f, 0.f, 0.f, 0.f, 0.f, 0.f};

#pragma unroll
  for (int tap = 0; tap < 3; ++tap)
#pragma unroll
    for (int kc = 0; kc < 2; ++kc) {
      v16h b0 = load_frag(&sm.a.bdil[tap][0][0], 80,  0, kc * 32, lane);
      v16h b1 = load_frag(&sm.a.bdil[tap][0][0], 80, 16, kc * 32, lane);
      v16h b2 = load_frag(&sm.a.bdil[tap][0][0], 80, 32, kc * 32, lane);
      v16h b3 = load_frag(&sm.a.bdil[tap][0][0], 80, 48, kc * 32, lane);
      accz[0] = WMMA_F16(a_dil[tap][kc], b0, accz[0]);
      accz[1] = WMMA_F16(a_dil[tap][kc], b1, accz[1]);
      accz[2] = WMMA_F16(a_dil[tap][kc], b2, accz[2]);
      accz[3] = WMMA_F16(a_dil[tap][kc], b3, accz[3]);
    }
#pragma unroll
  for (int kc = 0; kc < 3; ++kc) {
    v16h b0 = load_frag(&sm.a.bmel[0][0], MELP,  0, kc * 32, lane);
    v16h b1 = load_frag(&sm.a.bmel[0][0], MELP, 16, kc * 32, lane);
    v16h b2 = load_frag(&sm.a.bmel[0][0], MELP, 32, kc * 32, lane);
    v16h b3 = load_frag(&sm.a.bmel[0][0], MELP, 48, kc * 32, lane);
    accz[0] = WMMA_F16(a_mel[kc], b0, accz[0]);
    accz[1] = WMMA_F16(a_mel[kc], b1, accz[1]);
    accz[2] = WMMA_F16(a_mel[kc], b2, accz[2]);
    accz[3] = WMMA_F16(a_mel[kc], b3, accz[3]);
  }
  __syncthreads();   // all LDS reads done -> safe to overwrite union with z

  // ---- spill z (+fused dil/mel bias) to LDS -------------------------------
#pragma unroll
  for (int j = 0; j < 4; ++j) {
    int n = j * 16 + (lane & 15);
#pragma unroll
    for (int r = 0; r < 8; ++r) {
      int oc = mbase + r + ((lane >> 4) << 3);
      sm.b.z[oc][n] = accz[j][r] + bz[oc];
    }
  }
  __syncthreads();

  // ---- gating: sigmoid(gate) * tanh(sig), emit f16 [n][k] -----------------
  for (int idx = tid; idx < 64 * 64; idx += 256) {
    int cch = idx >> 6, n = idx & 63;
    float g = sm.b.z[cch][n];
    float s = sm.b.z[cch + 64][n];
    float zz = (1.0f / (1.0f + __expf(-g))) * tanhf(s);
    sm.b.bspl[n][cch] = (_Float16)zz;
  }
  __syncthreads();

  // ---- split conv GEMM + epilogue -----------------------------------------
  v16h a_spl[2];
#pragma unroll
  for (int kc = 0; kc < 2; ++kc)
    a_spl[kc] = load_frag(wspl, 64, mbase, kc * 32, lane);

  v8f acc2[4];
#pragma unroll
  for (int j = 0; j < 4; ++j)
    acc2[j] = (v8f){0.f, 0.f, 0.f, 0.f, 0.f, 0.f, 0.f, 0.f};
#pragma unroll
  for (int kc = 0; kc < 2; ++kc) {
    v16h b0 = load_frag(&sm.b.bspl[0][0], 80,  0, kc * 32, lane);
    v16h b1 = load_frag(&sm.b.bspl[0][0], 80, 16, kc * 32, lane);
    v16h b2 = load_frag(&sm.b.bspl[0][0], 80, 32, kc * 32, lane);
    v16h b3 = load_frag(&sm.b.bspl[0][0], 80, 48, kc * 32, lane);
    acc2[0] = WMMA_F16(a_spl[kc], b0, acc2[0]);
    acc2[1] = WMMA_F16(a_spl[kc], b1, acc2[1]);
    acc2[2] = WMMA_F16(a_spl[kc], b2, acc2[2]);
    acc2[3] = WMMA_F16(a_spl[kc], b3, acc2[3]);
  }

  const float rs2 = 0.70710678118654752f;
#pragma unroll
  for (int j = 0; j < 4; ++j) {
    int n = j * 16 + (lane & 15);
    int t = t0 + n;
#pragma unroll
    for (int r = 0; r < 8; ++r) {
      int oc = mbase + r + ((lane >> 4) << 3);
      float v = acc2[j][r] + spl_b[oc];
      if (oc < 64) {                                   // "out" half -> residual
        size_t off = ((size_t)b * CC + oc) * TT + t;
        h_out[off] = (h_in[off] + v) * rs2;
      } else {                                         // "skip" half -> accumulate
        size_t off = ((size_t)b * CC + (oc - 64)) * TT + t;
        skip[off] += v;
      }
    }
  }
}

// ---------------------------------------------------------------------------
// Time embedding + fc1 + fc2 + per-layer tfc projection (tiny; one block).
// ---------------------------------------------------------------------------
__global__ __launch_bounds__(256)
void time_mlp_kernel(const float* __restrict__ timev,
                     const float* __restrict__ fc1_w, const float* __restrict__ fc1_b,
                     const float* __restrict__ fc2_w, const float* __restrict__ fc2_b,
                     const float* __restrict__ tfc_w, const float* __restrict__ tfc_b,
                     float* __restrict__ t_all)                   // [D][B][64]
{
  __shared__ float tin[BB][128];
  __shared__ float h1[BB][512];
  __shared__ float h2[BB][512];
  const int tid = threadIdx.x;

  for (int idx = tid; idx < BB * 128; idx += 256) {
    int b = idx >> 7, j = idx & 127;
    int jj = j & 63;
    float ph = 100.0f * timev[b] * __powf(10.0f, (float)jj * (4.0f / 63.0f));
    tin[b][j] = (j < 64) ? __sinf(ph) : __cosf(ph);
  }
  __syncthreads();
  for (int idx = tid; idx < BB * 512; idx += 256) {
    int b = idx >> 9, o = idx & 511;
    float acc = fc1_b[o];
    for (int j = 0; j < 128; ++j) acc += tin[b][j] * fc1_w[o * 128 + j];
    h1[b][o] = acc / (1.0f + __expf(-acc));            // silu
  }
  __syncthreads();
  for (int idx = tid; idx < BB * 512; idx += 256) {
    int b = idx >> 9, o = idx & 511;
    float acc = fc2_b[o];
    for (int j = 0; j < 512; ++j) acc += h1[b][j] * fc2_w[o * 512 + j];
    h2[b][o] = acc / (1.0f + __expf(-acc));
  }
  __syncthreads();
  for (int idx = tid; idx < DD * BB * 64; idx += 256) {
    int dd = idx / 512, rem = idx % 512;
    int b = rem >> 6, cch = rem & 63;
    float acc = tfc_b[dd * 64 + cch];
    const float* w = tfc_w + (size_t)(dd * 64 + cch) * 512;
    for (int j = 0; j < 512; ++j) acc += h2[b][j] * w[j];
    t_all[idx] = acc;
  }
}

// ---------------------------------------------------------------------------
// Mel transposed-conv upsampling (x16 twice). Each output taps 3x2 weights.
// ---------------------------------------------------------------------------
__global__ void upsample1_kernel(const float* __restrict__ mel,   // [8][80][64]
                                 const float* __restrict__ w,     // [3][32]
                                 const float* __restrict__ bptr,
                                 float* __restrict__ m1)          // [8][80][1024]
{
  const float b0 = bptr[0];
  const size_t total = (size_t)BB * MELC * 1024;
  const size_t stride = (size_t)gridDim.x * blockDim.x;
  for (size_t idx = (size_t)blockIdx.x * blockDim.x + threadIdx.x; idx < total; idx += stride) {
    int wpos = (int)(idx % 1024);
    size_t rest = idx / 1024;
    int h = (int)(rest % MELC);
    int b = (int)(rest / MELC);
    float acc = b0;
    int kw0 = ((23 - wpos) % 16 + 16) % 16;
    for (int kh = 0; kh < 3; ++kh) {
      int hh = h + kh - 1;
      if (hh < 0 || hh >= MELC) continue;
      for (int kk = 0; kk < 2; ++kk) {
        int kw = kw0 + kk * 16;
        int num = wpos + kw - 23;
        if (num < 0) continue;
        int wi = num >> 4;
        if (wi >= 64) continue;
        acc += mel[((size_t)b * MELC + hh) * 64 + wi] * w[(2 - kh) * 32 + (31 - kw)];
      }
    }
    m1[idx] = leaky04(acc);
  }
}

// Second stage writes [b][t][96] (time-major, channel-contiguous, zero-padded)
// so each 64-step tile is one contiguous 12KB block for the TDM.
__global__ void upsample2_kernel(const float* __restrict__ m1,    // [8][80][1024]
                                 const float* __restrict__ w,
                                 const float* __restrict__ bptr,
                                 _Float16* __restrict__ m_bf)     // [8][16384][96]
{
  const float b0 = bptr[0];
  const size_t total = (size_t)BB * TT * MELP;
  const size_t stride = (size_t)gridDim.x * blockDim.x;
  for (size_t idx = (size_t)blockIdx.x * blockDim.x + threadIdx.x; idx < total; idx += stride) {
    int k = (int)(idx % MELP);
    size_t rest = idx / MELP;
    int t = (int)(rest % TT);
    int b = (int)(rest / TT);
    float outv = 0.f;
    if (k < MELC) {
      float acc = b0;
      int kw0 = ((23 - t) % 16 + 16) % 16;
      for (int kh = 0; kh < 3; ++kh) {
        int hh = k + kh - 1;
        if (hh < 0 || hh >= MELC) continue;
        for (int kk = 0; kk < 2; ++kk) {
          int kw = kw0 + kk * 16;
          int num = t + kw - 23;
          if (num < 0) continue;
          int wi = num >> 4;
          if (wi >= 1024) continue;
          acc += m1[((size_t)b * MELC + hh) * 1024 + wi] * w[(2 - kh) * 32 + (31 - kw)];
        }
      }
      outv = leaky04(acc);
    }
    m_bf[idx] = (_Float16)outv;
  }
}

// ---------------------------------------------------------------------------
// Input 1x1 conv (1 -> 64) + relu.
// ---------------------------------------------------------------------------
__global__ void cin_kernel(const float* __restrict__ x, const float* __restrict__ cw,
                           const float* __restrict__ cb, float* __restrict__ h)
{
  const size_t total = (size_t)BB * CC * TT;
  const size_t stride = (size_t)gridDim.x * blockDim.x;
  for (size_t idx = (size_t)blockIdx.x * blockDim.x + threadIdx.x; idx < total; idx += stride) {
    int t = (int)(idx % TT);
    size_t rest = idx / TT;
    int cch = (int)(rest % CC);
    int b   = (int)(rest / CC);
    float v = cw[cch] * x[(size_t)b * TT + t] + cb[cch];
    h[idx] = fmaxf(v, 0.f);
  }
}

__global__ void zero_kernel(float* __restrict__ p, size_t n)
{
  const size_t stride = (size_t)gridDim.x * blockDim.x;
  for (size_t idx = (size_t)blockIdx.x * blockDim.x + threadIdx.x; idx < n; idx += stride)
    p[idx] = 0.f;
}

// ---------------------------------------------------------------------------
// One-time weight conversion f32 -> f16 (layout transpose for dil conv),
// plus fused bias bz = dil_b + melc_b.
// ---------------------------------------------------------------------------
__global__ void prep_weights_kernel(const float* __restrict__ dil_w,   // [D][128][64][3]
                                    const float* __restrict__ dil_b,   // [D][128]
                                    const float* __restrict__ melc_w,  // [D][128][80]
                                    const float* __restrict__ melc_b,  // [D][128]
                                    const float* __restrict__ spl_w,   // [D][128][64]
                                    _Float16* __restrict__ wdil,       // [D][3][128][64]
                                    _Float16* __restrict__ wmel,       // [D][128][96]
                                    _Float16* __restrict__ wspl,       // [D][128][64]
                                    float* __restrict__ bz)            // [D][128]
{
  const size_t stride = (size_t)gridDim.x * blockDim.x;
  const size_t t0 = (size_t)blockIdx.x * blockDim.x + threadIdx.x;

  const size_t N1 = (size_t)DD * 3 * 128 * 64;
  for (size_t i = t0; i < N1; i += stride) {
    int ic = (int)(i & 63);
    size_t r = i >> 6;
    int oc = (int)(r & 127); r >>= 7;
    int tap = (int)(r % 3);
    int l   = (int)(r / 3);
    wdil[i] = (_Float16)dil_w[(((size_t)l * 128 + oc) * 64 + ic) * 3 + tap];
  }
  const size_t N2 = (size_t)DD * 128 * MELP;
  for (size_t i = t0; i < N2; i += stride) {
    int k = (int)(i % MELP);
    size_t r = i / MELP;
    int oc = (int)(r & 127);
    int l  = (int)(r >> 7);
    wmel[i] = (_Float16)((k < MELC) ? melc_w[((size_t)l * 128 + oc) * MELC + k] : 0.f);
  }
  const size_t N3 = (size_t)DD * 128 * 64;
  for (size_t i = t0; i < N3; i += stride)
    wspl[i] = (_Float16)spl_w[i];
  const size_t N4 = (size_t)DD * 128;
  for (size_t i = t0; i < N4; i += stride)
    bz[i] = dil_b[i] + melc_b[i];
}

// ---------------------------------------------------------------------------
// Final: relu(1x1 conv 64->64) then 1x1 conv 64->1, with skip pre-scale.
// ---------------------------------------------------------------------------
__global__ __launch_bounds__(256)
void final_kernel(const float* __restrict__ skip,
                  const float* __restrict__ f1w, const float* __restrict__ f1b,
                  const float* __restrict__ f2w, const float* __restrict__ f2b,
                  float* __restrict__ out)
{
  __shared__ float w1[64][64];
  __shared__ float w2[64];
  for (int idx = threadIdx.x; idx < 64 * 64; idx += 256) w1[idx >> 6][idx & 63] = f1w[idx];
  if (threadIdx.x < 64) w2[threadIdx.x] = f2w[threadIdx.x];
  __syncthreads();

  const float sc = 0.18257418583505537f;   // 1/sqrt(30)
  const size_t total = (size_t)BB * TT;
  const size_t stride = (size_t)gridDim.x * blockDim.x;
  for (size_t idx = (size_t)blockIdx.x * blockDim.x + threadIdx.x; idx < total; idx += stride) {
    int t = (int)(idx % TT);
    int b = (int)(idx / TT);
    float sk[64];
#pragma unroll
    for (int k = 0; k < 64; ++k) sk[k] = skip[((size_t)b * CC + k) * TT + t] * sc;
    float o = f2b[0];
    for (int cch = 0; cch < 64; ++cch) {
      float acc = f1b[cch];
#pragma unroll
      for (int k = 0; k < 64; ++k) acc += w1[cch][k] * sk[k];
      o += w2[cch] * fmaxf(acc, 0.f);
    }
    out[idx] = o;
  }
}

// ---------------------------------------------------------------------------
// Host launch
// ---------------------------------------------------------------------------
extern "C" void kernel_launch(void* const* d_in, const int* in_sizes, int n_in,
                              void* d_out, int out_size, void* d_ws, size_t ws_size,
                              hipStream_t stream)
{
  (void)in_sizes; (void)n_in; (void)out_size; (void)ws_size;

  const float* x      = (const float*)d_in[0];
  const float* mel    = (const float*)d_in[1];
  const float* timev  = (const float*)d_in[2];
  const float* fc1_w  = (const float*)d_in[3];
  const float* fc1_b  = (const float*)d_in[4];
  const float* fc2_w  = (const float*)d_in[5];
  const float* fc2_b  = (const float*)d_in[6];
  const float* tfc_w  = (const float*)d_in[7];
  const float* tfc_b  = (const float*)d_in[8];
  const float* cm1_w  = (const float*)d_in[9];
  const float* cm1_b  = (const float*)d_in[10];
  const float* cm2_w  = (const float*)d_in[11];
  const float* cm2_b  = (const float*)d_in[12];
  const float* cin_w  = (const float*)d_in[13];
  const float* cin_b  = (const float*)d_in[14];
  const float* dil_w  = (const float*)d_in[15];
  const float* dil_b  = (const float*)d_in[16];
  const float* melc_w = (const float*)d_in[17];
  const float* melc_b = (const float*)d_in[18];
  const float* spl_w  = (const float*)d_in[19];
  const float* spl_b  = (const float*)d_in[20];
  const float* f1_w   = (const float*)d_in[21];
  const float* f1_b   = (const float*)d_in[22];
  const float* f2_w   = (const float*)d_in[23];
  const float* f2_b   = (const float*)d_in[24];

  char* ws = (char*)d_ws;
  auto carve = [&](size_t bytes) -> char* {
    char* p = ws;
    ws += (bytes + 255) & ~(size_t)255;
    return p;
  };
  float*    t_all = (float*)carve((size_t)DD * BB * 64 * sizeof(float));
  float*    m1    = (float*)carve((size_t)BB * MELC * 1024 * sizeof(float));
  _Float16* m_bf  = (_Float16*)carve((size_t)BB * TT * MELP * sizeof(_Float16));
  float*    h_a   = (float*)carve((size_t)BB * CC * TT * sizeof(float));
  float*    h_b   = (float*)carve((size_t)BB * CC * TT * sizeof(float));
  float*    skip  = (float*)carve((size_t)BB * CC * TT * sizeof(float));
  _Float16* wdil  = (_Float16*)carve((size_t)DD * 3 * 128 * 64 * sizeof(_Float16));
  _Float16* wmel  = (_Float16*)carve((size_t)DD * 128 * MELP * sizeof(_Float16));
  _Float16* wspl  = (_Float16*)carve((size_t)DD * 128 * 64 * sizeof(_Float16));
  float*    bz    = (float*)carve((size_t)DD * 128 * sizeof(float));

  prep_weights_kernel<<<256, 256, 0, stream>>>(dil_w, dil_b, melc_w, melc_b, spl_w,
                                               wdil, wmel, wspl, bz);
  time_mlp_kernel<<<1, 256, 0, stream>>>(timev, fc1_w, fc1_b, fc2_w, fc2_b,
                                         tfc_w, tfc_b, t_all);
  upsample1_kernel<<<640, 256, 0, stream>>>(mel, cm1_w, cm1_b, m1);
  upsample2_kernel<<<2048, 256, 0, stream>>>(m1, cm2_w, cm2_b, m_bf);
  cin_kernel<<<2048, 256, 0, stream>>>(x, cin_w, cin_b, h_a);
  zero_kernel<<<2048, 256, 0, stream>>>(skip, (size_t)BB * CC * TT);

  dim3 grid(TT / 64, BB);
  for (int i = 0; i < DD; ++i) {
    int d = 1 << (i % 10);
    const float* hin = (i & 1) ? h_b : h_a;
    float*      hout = (i & 1) ? h_a : h_b;
    layer_kernel<<<grid, 256, 0, stream>>>(hin, hout, skip, m_bf,
                                           t_all + (size_t)i * BB * 64,
                                           wdil  + (size_t)i * 3 * 128 * 64,
                                           wmel  + (size_t)i * 128 * MELP,
                                           wspl  + (size_t)i * 128 * 64,
                                           bz    + (size_t)i * 128,
                                           spl_b + (size_t)i * 128,
                                           d);
  }
  final_kernel<<<512, 256, 0, stream>>>(skip, f1_w, f1_b, f2_w, f2_b, (float*)d_out);
}
